// Retention_49675591745550
// MI455X (gfx1250) — matrix operand order repaired and checked by reading, not compiled, not run on hardware
//
#include <hip/hip_runtime.h>
#include <hip/hip_bf16.h>

// Problem constants (match reference)
#define kT 4096
#define kH 8
#define kD 64
#define kL 1024
#define kC 4
#define kNN 80          // padded N: 64 value cols + 1 ones/denominator col + 15 zero
#define kKP 4096        // expanded feature dim D*D
#define SCALE_F 0.125f
#define EPS_F 1e-6f

typedef __attribute__((ext_vector_type(16))) __bf16 v16bf;
typedef __attribute__((ext_vector_type(8)))  float  v8f;

struct bfrag16 { union { v16bf v; uint4 u[2]; }; };

// ---- bf16x3 WMMA: C += Ahi*Bhi + Ahi*Blo + Alo*Bhi (near-fp32 GEMM) ----
__device__ __forceinline__ v8f mma3(v16bf ah, v16bf al, v16bf bh, v16bf bl, v8f c) {
  c = __builtin_amdgcn_wmma_f32_16x16x32_bf16(false, ah, false, bh, (short)0, c, false, false);
  c = __builtin_amdgcn_wmma_f32_16x16x32_bf16(false, ah, false, bl, (short)0, c, false, false);
  c = __builtin_amdgcn_wmma_f32_16x16x32_bf16(false, al, false, bh, (short)0, c, false, false);
  return c;
}

// A-frag from pre-split row-major panel: rowp=&panel[row*ld]; k = 8h+(0..7), 16+8h+(0..7)
__device__ __forceinline__ v16bf load_a_bf16(const __bf16* rowp, int half) {
  bfrag16 f;
  f.u[0] = *(const uint4*)(rowp + 8 * half);
  f.u[1] = *(const uint4*)(rowp + 16 + 8 * half);
  return f.v;
}
// B-frag from pre-split N-major (transposed) panel: colp=&panel[col*ld]; k = 8h+(0..7), 16+8h+(0..7)
__device__ __forceinline__ v16bf load_b_bf16(const __bf16* colp, int half) {
  bfrag16 f;
  f.u[0] = *(const uint4*)(colp + 8 * half);
  f.u[1] = *(const uint4*)(colp + 16 + 8 * half);
  return f.v;
}
// build phi-style A frag: a(row,k) = s * src[k]
__device__ __forceinline__ void build_a_scaled(const float* src, float s, int half,
                                               v16bf& ah, v16bf& al) {
#pragma unroll
  for (int j = 0; j < 16; ++j) {
    int kk = ((j >> 3) << 4) + (half << 3) + (j & 7);
    float x = s * src[kk];
    __bf16 hb = (__bf16)x;
    ah[j] = hb;  al[j] = (__bf16)(x - (float)hb);
  }
}

// ---------------- gate cumsum: g, gend, wdec = exp(g_end - g) ----------------
__global__ void pr_gate_kernel(const float* __restrict__ log_g, float* __restrict__ g,
                               float* __restrict__ gend, float* __restrict__ wdec) {
  int t = threadIdx.x;
  if (t >= kC * kH) return;
  int c = t >> 3, h = t & 7;
  size_t chb = (size_t)c * kH + h;
  float s = 0.f;
  for (int l = 0; l < kL; ++l) {
    s += log_g[(size_t)(c * kL + l) * kH + h];
    g[chb * kL + l] = s;
  }
  gend[chb] = s;
  for (int l = 0; l < kL; ++l) wdec[chb * kL + l] = __expf(s - g[chb * kL + l]);
}

// ---------------- one-shot pre-split of q (scaled) and k; k transpose copies ----------------
__global__ void pr_split_qk(const float* __restrict__ q, const float* __restrict__ k,
                            const float* __restrict__ wdec,
                            __bf16* __restrict__ qh, __bf16* __restrict__ ql,
                            __bf16* __restrict__ kh, __bf16* __restrict__ kl,
                            float* __restrict__ ktr, float* __restrict__ kwtr) {
  int idx = blockIdx.x * blockDim.x + threadIdx.x;
  if (idx >= kC * kH * kL * kD) return;
  int d = idx & 63, l = (idx >> 6) & (kL - 1), h = (idx >> 16) & 7, c = idx >> 19;
  size_t src = (((size_t)(c * kL + l) * kH + h) * kD) + d;
  float qv = SCALE_F * q[src];
  __bf16 qhb = (__bf16)qv;
  qh[idx] = qhb;  ql[idx] = (__bf16)(qv - (float)qhb);
  float kv = k[src];
  __bf16 khb = (__bf16)kv;
  kh[idx] = khb;  kl[idx] = (__bf16)(kv - (float)khb);
  size_t chb = (size_t)c * kH + h;
  size_t tidx = (chb * kD + d) * kL + l;
  ktr[tidx] = kv;
  kwtr[tidx] = kv * wdec[chb * kL + l];
}

// ---------------- one-shot pre-split of [V | 1 | 0], transposed [n][l] ----------------
__global__ void pr_split_v(const float* __restrict__ v,
                           __bf16* __restrict__ vth, __bf16* __restrict__ vtl) {
  int idx = blockIdx.x * blockDim.x + threadIdx.x;
  if (idx >= kC * kH * kNN * kL) return;
  int l = idx & (kL - 1);
  int n = (idx >> 10) % kNN;
  int ch = idx / (kNN * kL);
  int h = ch & 7, c = ch >> 3;
  float val = (n < kD) ? v[(((size_t)(c * kL + l) * kH + h) * kD) + n] : (n == kD ? 1.f : 0.f);
  __bf16 hb = (__bf16)val;
  vth[idx] = hb;  vtl[idx] = (__bf16)(val - (float)hb);
}

__global__ void pr_zero_u32(unsigned int* __restrict__ p, size_t n) {
  size_t i = (size_t)blockIdx.x * blockDim.x + threadIdx.x;
  size_t stride = (size_t)gridDim.x * blockDim.x;
  for (; i < n; i += stride) p[i] = 0u;
}

// ---------------- inter: interbuf[h][l][0:80] = phi(q) @ S ----------------
// Wave owns 2 x 16-row strips sharing every B-fragment load; barrier-free.
__global__ void __launch_bounds__(128)
pr_inter_kernel(const float* __restrict__ q, const __bf16* __restrict__ Sth,
                const __bf16* __restrict__ Stl, float* __restrict__ interbuf, int c) {
  int tid = threadIdx.x, lane = tid & 31, w = tid >> 5;
  int m0 = blockIdx.x * 128, h = blockIdx.y;
  int colc = lane & 15, half = lane >> 4;
  int row0 = m0 + w * 32 + colc;
  const float* qrow0 = q + (((size_t)(c * kL + row0) * kH + h) * kD);
  const float* qrow1 = qrow0 + (size_t)16 * kH * kD;
  const __bf16* sh0 = Sth + (size_t)h * kNN * kKP;
  const __bf16* sl0 = Stl + (size_t)h * kNN * kKP;
  const float S2 = SCALE_F * SCALE_F;

  v8f acc[2][5] = {};
  for (int kt = 0; kt < kKP / 32; ++kt) {
    int d = kt >> 1, e0 = (kt & 1) << 5;
    v16bf ah0, al0, ah1, al1;
    build_a_scaled(qrow0 + e0, S2 * qrow0[d], half, ah0, al0);
    build_a_scaled(qrow1 + e0, S2 * qrow1[d], half, ah1, al1);
#pragma unroll
    for (int n = 0; n < 5; ++n) {
      size_t cb = (size_t)(n * 16 + colc) * kKP + kt * 32;
      v16bf bh = load_b_bf16(sh0 + cb, half);
      v16bf bl = load_b_bf16(sl0 + cb, half);
      acc[0][n] = mma3(ah0, al0, bh, bl, acc[0][n]);
      acc[1][n] = mma3(ah1, al1, bh, bl, acc[1][n]);
    }
  }
#pragma unroll
  for (int s = 0; s < 2; ++s)
#pragma unroll
    for (int n = 0; n < 5; ++n)
#pragma unroll
      for (int r = 0; r < 8; ++r) {
        int rr = m0 + w * 32 + s * 16 + r + half * 8;
        interbuf[((size_t)h * kL + rr) * kNN + n * 16 + colc] = acc[s][n][r];
      }
}

// ---------------- intra: causal gated-square attention, pre-split operands ----------------
__global__ void __launch_bounds__(128)
pr_intra_kernel(const __bf16* __restrict__ qh, const __bf16* __restrict__ ql,
                const __bf16* __restrict__ kh, const __bf16* __restrict__ kl,
                const __bf16* __restrict__ vth, const __bf16* __restrict__ vtl,
                const float* __restrict__ g, float* __restrict__ intrabuf, int c) {
  __shared__ __align__(16) __bf16 pbh[4][16 * 32];
  __shared__ __align__(16) __bf16 pbl[4][16 * 32];
  int tid = threadIdx.x, lane = tid & 31, w = tid >> 5;
  int m0 = blockIdx.x * 64, h = blockIdx.y;
  int colc = lane & 15, half = lane >> 4;
  size_t chb = (size_t)c * kH + h;
  const __bf16* qhB = qh + chb * kL * kD;
  const __bf16* qlB = ql + chb * kL * kD;
  const __bf16* khB = kh + chb * kL * kD;
  const __bf16* klB = kl + chb * kL * kD;
  const __bf16* vthB = vth + chb * kNN * kL;
  const __bf16* vtlB = vtl + chb * kNN * kL;
  const float* gch = g + chb * kL;

  float egr[8];                                        // exp(g_row), loop-invariant
#pragma unroll
  for (int r = 0; r < 8; ++r) egr[r] = __expf(gch[m0 + w * 16 + half * 8 + r]);
  const __bf16* qhrow = qhB + (size_t)(m0 + w * 16 + colc) * kD;
  const __bf16* qlrow = qlB + (size_t)(m0 + w * 16 + colc) * kD;

  v8f acc[5] = {};
  int nks = (m0 + 64) >> 5;
  for (int ks = 0; ks < nks; ++ks) {
    int s0 = ks * 32;
    // scores: qs(16x64) @ k^T(64x32)
    v8f sc[2] = {};
#pragma unroll
    for (int kk2 = 0; kk2 < 2; ++kk2) {
      v16bf ah = load_a_bf16(qhrow + kk2 * 32, half);
      v16bf al = load_a_bf16(qlrow + kk2 * 32, half);
#pragma unroll
      for (int nt = 0; nt < 2; ++nt) {
        size_t cb = (size_t)(s0 + nt * 16 + colc) * kD + kk2 * 32;
        sc[nt] = mma3(ah, al, load_b_bf16(khB + cb, half), load_b_bf16(klB + cb, half), sc[nt]);
      }
    }
    // mask + gate + square, split to bf16 P tile in LDS (per-wave region)
    float ek0 = __expf(-gch[s0 + colc]);
    float ek1 = __expf(-gch[s0 + 16 + colc]);
#pragma unroll
    for (int nt = 0; nt < 2; ++nt) {
      float ekn = nt ? ek1 : ek0;
      int sl = nt * 16 + colc;
      int skey = s0 + sl;
#pragma unroll
      for (int r = 0; r < 8; ++r) {
        int lrow = m0 + w * 16 + r + half * 8;
        float t = sc[nt][r];
        float a = (skey <= lrow) ? (egr[r] * ekn) * (t * t) : 0.f;
        __bf16 hb = (__bf16)a;
        pbh[w][(r + half * 8) * 32 + sl] = hb;
        pbl[w][(r + half * 8) * 32 + sl] = (__bf16)(a - (float)hb);
      }
    }
    // O += P(16x32) @ [V|1|0](32x80)
    v16bf ah = load_a_bf16(&pbh[w][colc * 32], half);
    v16bf al = load_a_bf16(&pbl[w][colc * 32], half);
#pragma unroll
    for (int n = 0; n < 5; ++n) {
      size_t cb = (size_t)(n * 16 + colc) * kL + s0;
      acc[n] = mma3(ah, al, load_b_bf16(vthB + cb, half), load_b_bf16(vtlB + cb, half), acc[n]);
    }
  }
#pragma unroll
  for (int n = 0; n < 5; ++n)
#pragma unroll
    for (int r = 0; r < 8; ++r) {
      int rr = m0 + w * 16 + r + half * 8;
      intrabuf[((size_t)h * kL + rr) * kNN + n * 16 + colc] = acc[n][r];
    }
}

// ---------------- combine: out = (intra + e^g * inter_raw) / max(den, eps) ----------------
__global__ void pr_combine_kernel(const float* __restrict__ interbuf,
                                  const float* __restrict__ intrabuf,
                                  const float* __restrict__ g,
                                  float* __restrict__ out, int c) {
  int idx = blockIdx.x * blockDim.x + threadIdx.x;
  if (idx >= kL * kH * kD) return;
  int d = idx % kD;
  int h = (idx / kD) % kH;
  int l = idx / (kD * kH);
  float eg = __expf(g[((size_t)c * kH + h) * kL + l]);
  size_t base = ((size_t)h * kL + l) * kNN;
  float num = intrabuf[base + d] + eg * interbuf[base + d];
  float den = intrabuf[base + kD] + eg * interbuf[base + kD];
  den = fmaxf(den, EPS_F);
  out[(((size_t)(c * kL + l)) * kH + h) * kD + d] = num / den;
}

// ---------------- state update: S[n][d*64+e] = gt*S + (kc_e ⊙ kw_d)^T @ [V|1|0] ----------------
// 2 waves; each wave owns 2 x 16-row e-strips sharing every B-fragment load.
__global__ void __launch_bounds__(64)
pr_update_kernel(const float* __restrict__ ktr, const float* __restrict__ kwtr,
                 const __bf16* __restrict__ vth, const __bf16* __restrict__ vtl,
                 const float* __restrict__ gend,
                 __bf16* __restrict__ Sth, __bf16* __restrict__ Stl, int c) {
  int tid = threadIdx.x, lane = tid & 31, w = tid >> 5;
  int d = blockIdx.x, h = blockIdx.y;
  int colc = lane & 15, half = lane >> 4;
  size_t chb = (size_t)c * kH + h;
  const float* ke_row0 = ktr + (chb * kD + (w * 32 + colc)) * kL;   // kc[:, e] transposed
  const float* ke_row1 = ke_row0 + (size_t)16 * kL;
  const float* kwd_row = kwtr + (chb * kD + d) * kL;                // kc[:, d]*exp(ge-g)
  const __bf16* vthB = vth + chb * kNN * kL;
  const __bf16* vtlB = vtl + chb * kNN * kL;

  v8f acc[2][5] = {};
  for (int ks = 0; ks < kL / 32; ++ks) {
    int l0 = ks * 32;
    v16bf ah0, al0, ah1, al1;
#pragma unroll
    for (int j = 0; j < 16; ++j) {                    // per-step split: product kc_e*kw_d
      int kk = ((j >> 3) << 4) + (half << 3) + (j & 7);
      float wv = kwd_row[l0 + kk];
      float x0 = ke_row0[l0 + kk] * wv;
      float x1 = ke_row1[l0 + kk] * wv;
      __bf16 h0 = (__bf16)x0;
      ah0[j] = h0;  al0[j] = (__bf16)(x0 - (float)h0);
      __bf16 h1 = (__bf16)x1;
      ah1[j] = h1;  al1[j] = (__bf16)(x1 - (float)h1);
    }
#pragma unroll
    for (int n = 0; n < 5; ++n) {
      size_t cb = (size_t)(n * 16 + colc) * kL + l0;
      v16bf bh = load_b_bf16(vthB + cb, half);
      v16bf bl = load_b_bf16(vtlB + cb, half);
      acc[0][n] = mma3(ah0, al0, bh, bl, acc[0][n]);
      acc[1][n] = mma3(ah1, al1, bh, bl, acc[1][n]);
    }
  }
  float gt = __expf(gend[c * kH + h]);
#pragma unroll
  for (int s = 0; s < 2; ++s)
#pragma unroll
    for (int n = 0; n < 5; ++n)
#pragma unroll
      for (int r = 0; r < 8; ++r) {
        int e = w * 32 + s * 16 + r + half * 8;
        size_t idx = ((size_t)h * kNN + n * 16 + colc) * kKP + (size_t)d * 64 + e;
        float oldv = (float)Sth[idx] + (float)Stl[idx];  // f32 reconstruction
        float nv = gt * oldv + acc[s][n][r];
        __bf16 hb = (__bf16)nv;
        Sth[idx] = hb;  Stl[idx] = (__bf16)(nv - (float)hb);
      }
}

extern "C" void kernel_launch(void* const* d_in, const int* in_sizes, int n_in,
                              void* d_out, int out_size, void* d_ws, size_t ws_size,
                              hipStream_t stream) {
  const float* q  = (const float*)d_in[0];
  const float* k  = (const float*)d_in[1];
  const float* v  = (const float*)d_in[2];
  const float* lg = (const float*)d_in[3];
  float* out = (float*)d_out;

  char* p = (char*)d_ws;
  auto carve = [&](size_t bytes) { void* r = (void*)p; p += (bytes + 255) & ~(size_t)255; return r; };
  float* g        = (float*)carve((size_t)kC * kH * kL * 4);
  float* gend     = (float*)carve((size_t)kC * kH * 4);
  float* wdec     = (float*)carve((size_t)kC * kH * kL * 4);
  float* interbuf = (float*)carve((size_t)kH * kL * kNN * 4);
  float* intrabuf = (float*)carve((size_t)kH * kL * kNN * 4);
  __bf16* qh      = (__bf16*)carve((size_t)kC * kH * kL * kD * 2);
  __bf16* ql      = (__bf16*)carve((size_t)kC * kH * kL * kD * 2);
  __bf16* kh      = (__bf16*)carve((size_t)kC * kH * kL * kD * 2);
  __bf16* kl      = (__bf16*)carve((size_t)kC * kH * kL * kD * 2);
  float*  ktr     = (float*)carve((size_t)kC * kH * kL * kD * 4);
  float*  kwtr    = (float*)carve((size_t)kC * kH * kL * kD * 4);
  __bf16* vth     = (__bf16*)carve((size_t)kC * kH * kNN * kL * 2);
  __bf16* vtl     = (__bf16*)carve((size_t)kC * kH * kNN * kL * 2);
  __bf16* Sth     = (__bf16*)carve((size_t)kH * kNN * kKP * 2);
  __bf16* Stl     = (__bf16*)carve((size_t)kH * kNN * kKP * 2);

  pr_gate_kernel<<<1, 32, 0, stream>>>(lg, g, gend, wdec);
  {
    int n = kC * kH * kL * kD;
    pr_split_qk<<<(n + 255) / 256, 256, 0, stream>>>(q, k, wdec, qh, ql, kh, kl, ktr, kwtr);
  }
  {
    int n = kC * kH * kNN * kL;
    pr_split_v<<<(n + 255) / 256, 256, 0, stream>>>(v, vth, vtl);
  }
  {
    size_t n32 = (size_t)kH * kNN * kKP / 2;   // zero both bf16 S arrays as u32
    pr_zero_u32<<<(int)((n32 + 255) / 256), 256, 0, stream>>>((unsigned int*)Sth, n32);
    pr_zero_u32<<<(int)((n32 + 255) / 256), 256, 0, stream>>>((unsigned int*)Stl, n32);
  }
  for (int c = 0; c < kC; ++c) {
    pr_inter_kernel<<<dim3(8, kH), 128, 0, stream>>>(q, Sth, Stl, interbuf, c);
    pr_intra_kernel<<<dim3(16, kH), 128, 0, stream>>>(qh, ql, kh, kl, vth, vtl, g, intrabuf, c);
    pr_combine_kernel<<<(kL * kH * kD + 255) / 256, 256, 0, stream>>>(interbuf, intrabuf, g, out, c);
    pr_update_kernel<<<dim3(64, kH), 64, 0, stream>>>(ktr, kwtr, vth, vtl, gend, Sth, Stl, c);
  }
}
